// Net_66597762891900
// MI455X (gfx1250) — matrix operand order repaired and checked by reading, not compiled, not run on hardware
//
#include <hip/hip_runtime.h>

#define NN   100000
#define EE   1600000
#define DD   128
#define HH   256
#define GG   64
#define NN0  50000
#define NN1  25000

typedef __attribute__((ext_vector_type(16))) __bf16 v16bf;
typedef __attribute__((ext_vector_type(8)))  float  v8f;

__device__ __forceinline__ float lrelu(float v) { return v > 0.f ? v : 0.01f * v; }

// ---------------------------------------------------------------------------
// Fused linear layer via WMMA:  C[M,NOUT] = act(((A[M,K] @ W[NOUT,K]^T) + b) * g + be)
// Block = 256 threads = 8 waves; each wave computes 16 rows x NOUT cols.
// W staged in LDS as bf16 (shared by all waves), A converted f32->bf16 on load.
// A loads are UNconditional (row clamped to M-1); out-of-range rows produce
// garbage accumulators whose stores are guarded -- keeps the load path as
// straight-line global_load_b128 + v_cvt_pk_bf16_f32, no exec-mask churn.
// ---------------------------------------------------------------------------
template<int NOUT>
__global__ __launch_bounds__(256)
void wmma_linear(const float* __restrict__ A, const float* __restrict__ W,
                 const float* __restrict__ bias, const float* __restrict__ gain,
                 const float* __restrict__ beta, float* __restrict__ C,
                 int M, int K, int act)
{
    constexpr int NT = NOUT / 16;
    __shared__ __align__(32) __bf16 Wlds[32768];                 // 64 KB
    const int KP = ((K + 16) * NOUT <= 32768) ? (K + 16) : K;    // pad rows vs bank conflicts when it fits

    for (int i = threadIdx.x; i < NOUT * K; i += 256) {
        int n = i / K, k = i - n * K;
        Wlds[n * KP + k] = (__bf16)W[i];
    }
    __syncthreads();

    const int  wave = threadIdx.x >> 5;
    const int  lane = threadIdx.x & 31;
    const int  l16  = lane & 15;
    const bool hi   = lane >= 16;
    const int  rowBase = blockIdx.x * 128 + wave * 16;

    const int  arow = rowBase + l16;
    const float* Arow = A + (size_t)(arow < M ? arow : (M - 1)) * K;
    const int  koff = hi ? 16 : 0;

    v8f acc[NT];
#pragma unroll
    for (int n = 0; n < NT; ++n) acc[n] = (v8f){0.f,0.f,0.f,0.f,0.f,0.f,0.f,0.f};

    for (int kb = 0; kb < K; kb += 32) {
        // A fragment (16x32 bf16): lane<16 -> K = kb+0..15; lane>=16 -> kb+16..31, contiguous
        const float4* p0 = (const float4*)(Arow + kb + koff);
        const float4* p1 = (const float4*)(Arow + kb + 8 + koff);
        const float4 q0 = p0[0];
        const float4 q1 = p0[1];
        const float4 q2 = p1[0];
        const float4 q3 = p1[1];
        v16bf a;
        a[0]=(__bf16)q0.x;  a[1]=(__bf16)q0.y;  a[2]=(__bf16)q0.z;  a[3]=(__bf16)q0.w;
        a[4]=(__bf16)q1.x;  a[5]=(__bf16)q1.y;  a[6]=(__bf16)q1.z;  a[7]=(__bf16)q1.w;
        a[8]=(__bf16)q2.x;  a[9]=(__bf16)q2.y;  a[10]=(__bf16)q2.z; a[11]=(__bf16)q2.w;
        a[12]=(__bf16)q3.x; a[13]=(__bf16)q3.y; a[14]=(__bf16)q3.z; a[15]=(__bf16)q3.w;
#pragma unroll
        for (int n = 0; n < NT; ++n) {
            // B fragment (32x16 bf16): lanes 0-15 -> K kb+0..15, lanes 16-31 -> kb+16..31, contiguous
            const v16bf b = *(const v16bf*)&Wlds[(n * 16 + l16) * KP + kb + (hi ? 16 : 0)];
            acc[n] = __builtin_amdgcn_wmma_f32_16x16x32_bf16(
                         false, a, false, b, (short)0, acc[n], false, false);
        }
    }

    const float slope = act ? 0.01f : 1.0f;   // branchless activation (slope 1 == identity)
    const int crowBase = rowBase + (hi ? 8 : 0);
#pragma unroll
    for (int n = 0; n < NT; ++n) {
        const int col = n * 16 + l16;
        const float bv = bias ? bias[col] : 0.f;
        const float gv = gain ? gain[col] : 1.f;
        const float ev = beta ? beta[col] : 0.f;
#pragma unroll
        for (int r = 0; r < 8; ++r) {
            const int row = crowBase + r;
            if (row < M) {
                float v = (acc[n][r] + bv) * gv + ev;
                v = (v > 0.f) ? v : slope * v;
                C[(size_t)row * NOUT + col] = v;
            }
        }
    }
}

// ---------------------------------------------------------------------------
// Graph / pooling helper kernels
// ---------------------------------------------------------------------------
__global__ void k_fill_f32(float* p, float v, long n) {
    long i = blockIdx.x * (long)blockDim.x + threadIdx.x;
    long s = (long)gridDim.x * blockDim.x;
    for (; i < n; i += s) p[i] = v;
}

__global__ void k_fill_i32(int* p, int v, long n) {
    long i = blockIdx.x * (long)blockDim.x + threadIdx.x;
    long s = (long)gridDim.x * blockDim.x;
    for (; i < n; i += s) p[i] = v;
}

__global__ void k_deg(const int* __restrict__ rows, float* __restrict__ deg, int E) {
    int e = blockIdx.x * blockDim.x + threadIdx.x;
    if (e < E) atomicAdd(&deg[rows[e]], 1.0f);
}

__global__ void k_rsqrt(float* p, int n) {
    int i = blockIdx.x * blockDim.x + threadIdx.x;
    if (i < n) p[i] = rsqrtf(p[i]);
}

// h[i] = dis[i]^2 * hmlp[i] (+ x[i] residual if x != null)   -- self-loop term + init
__global__ void k_self_init(const float* __restrict__ hm, const float* __restrict__ x,
                            const float* __restrict__ dis, float* __restrict__ h, int n) {
    long idx = blockIdx.x * (long)blockDim.x + threadIdx.x;
    if (idx >= (long)n * 32) return;
    int i = (int)(idx >> 5), q = (int)(idx & 31);
    float w = dis[i] * dis[i];
    float4 hv = ((const float4*)(hm + (size_t)i * DD))[q];
    float4 o = {w * hv.x, w * hv.y, w * hv.z, w * hv.w};
    if (x) {
        float4 xv = ((const float4*)(x + (size_t)i * DD))[q];
        o.x += xv.x; o.y += xv.y; o.z += xv.z; o.w += xv.w;
    }
    ((float4*)(h + (size_t)i * DD))[q] = o;
}

// one wave per edge: lane handles float4 at d = lane*4; 128-wide feature scatter-add
__global__ void k_edge_scatter(const int* __restrict__ row, const int* __restrict__ col,
                               const float* __restrict__ ea, const float* __restrict__ dis,
                               const float* __restrict__ hm, float* __restrict__ out, int E) {
    int e = (int)((blockIdx.x * 256u + threadIdx.x) >> 5);
    if (e >= E) return;
    int lane = threadIdx.x & 31;
    int r = row[e], c = col[e];
    float w = dis[r] * dis[c] * ea[e];
    float4 v = ((const float4*)(hm + (size_t)r * DD))[lane];
    float* dst = out + (size_t)c * DD + lane * 4;
    atomicAdd(dst + 0, w * v.x);
    atomicAdd(dst + 1, w * v.y);
    atomicAdd(dst + 2, w * v.z);
    atomicAdd(dst + 3, w * v.w);
}

__global__ void k_pool_max_f(const float* __restrict__ src, const int* __restrict__ cl,
                             float* __restrict__ dst, int n, int act) {
    long idx = blockIdx.x * (long)blockDim.x + threadIdx.x;
    if (idx >= (long)n * DD) return;
    int i = (int)(idx >> 7), d = (int)(idx & 127);
    float v = src[idx];
    if (act) v = lrelu(v);
    atomicMax(&dst[(size_t)cl[i] * DD + d], v);
}

__global__ void k_pool_max_i(const int* __restrict__ src, const int* __restrict__ cl,
                             int* __restrict__ dst, int n) {
    int i = blockIdx.x * blockDim.x + threadIdx.x;
    if (i < n) atomicMax(&dst[cl[i]], src[i]);
}

__global__ void k_combine(float* __restrict__ px, const float* __restrict__ ox, long n) {
    long i = blockIdx.x * (long)blockDim.x + threadIdx.x;
    if (i >= n) return;
    float a = px[i], b = ox[i];
    if (a == -__builtin_inff()) a = 0.f;
    if (b == -__builtin_inff()) b = 0.f;
    px[i] = a + b;
}

__global__ void k_fixneg(float* p, long n) {
    long i = blockIdx.x * (long)blockDim.x + threadIdx.x;
    if (i < n && p[i] == -__builtin_inff()) p[i] = 0.f;
}

__global__ void k_map_edges(const int* __restrict__ ei, const int* __restrict__ c0,
                            const int* __restrict__ c1, int* __restrict__ r2,
                            int* __restrict__ c2, int E) {
    int e = blockIdx.x * blockDim.x + threadIdx.x;
    if (e >= E) return;
    r2[e] = c1[c0[ei[e]]];
    c2[e] = c1[c0[ei[EE + e]]];
}

// pad lw1 [128,136] -> [128,160] with zeros (K multiple of 32 for WMMA)
__global__ void k_pad_w(const float* __restrict__ w, float* __restrict__ wp) {
    int idx = blockIdx.x * blockDim.x + threadIdx.x;
    if (idx >= 128 * 160) return;
    int r = idx / 160, c = idx - r * 160;
    wp[idx] = (c < 136) ? w[r * 136 + c] : 0.f;
}

// z[G,160] = [ fix(gx) | lrelu((energy@emb_w^T + eb)*eg + ebe) | zero-pad ]
__global__ void k_build_z(const float* __restrict__ gx, const float* __restrict__ en,
                          const float* __restrict__ ew, const float* __restrict__ eb,
                          const float* __restrict__ eg, const float* __restrict__ ebe,
                          float* __restrict__ z) {
    int idx = blockIdx.x * blockDim.x + threadIdx.x;
    if (idx >= GG * 160) return;
    int g = idx / 160, c = idx - g * 160;
    float v;
    if (c < 128) {
        v = gx[g * 128 + c];
        if (v == -__builtin_inff()) v = 0.f;
    } else if (c < 136) {
        int o = c - 128;
        float s = eb[o];
        for (int k = 0; k < 21; ++k) s += en[g * 21 + k] * ew[o * 21 + k];
        v = lrelu(s * eg[o] + ebe[o]);
    } else {
        v = 0.f;
    }
    z[idx] = v;
}

__global__ void k_head2(const float* __restrict__ zz, const float* __restrict__ w,
                        const float* __restrict__ b, float* __restrict__ out) {
    int g = threadIdx.x;
    if (g >= GG) return;
    float s = b[0];
    for (int d = 0; d < 128; ++d) s += zz[g * 128 + d] * w[d];
    out[g] = s;
}

// ---------------------------------------------------------------------------
extern "C" void kernel_launch(void* const* d_in, const int* in_sizes, int n_in,
                              void* d_out, int out_size, void* d_ws, size_t ws_size,
                              hipStream_t stream)
{
    (void)in_sizes; (void)n_in; (void)out_size; (void)ws_size;

    const float* x    = (const float*)d_in[0];
    const float* eatt = (const float*)d_in[1];
    const float* en   = (const float*)d_in[2];
    const float* w0a  = (const float*)d_in[3];
    const float* b0a  = (const float*)d_in[4];
    const float* g0   = (const float*)d_in[5];
    const float* be0  = (const float*)d_in[6];
    const float* w0b  = (const float*)d_in[7];
    const float* b0b  = (const float*)d_in[8];
    const float* w2a  = (const float*)d_in[15];
    const float* b2a  = (const float*)d_in[16];
    const float* g2   = (const float*)d_in[17];
    const float* be2  = (const float*)d_in[18];
    const float* w2b  = (const float*)d_in[19];
    const float* b2b  = (const float*)d_in[20];
    const float* embw = (const float*)d_in[21];
    const float* embb = (const float*)d_in[22];
    const float* embg = (const float*)d_in[23];
    const float* embe = (const float*)d_in[24];
    const float* lw1  = (const float*)d_in[25];
    const float* lb1  = (const float*)d_in[26];
    const float* lg   = (const float*)d_in[27];
    const float* lbe  = (const float*)d_in[28];
    const float* lw2  = (const float*)d_in[29];
    const float* lb2  = (const float*)d_in[30];
    const int*   ei    = (const int*)d_in[31];
    const int*   batch = (const int*)d_in[32];
    const int*   c0    = (const int*)d_in[33];
    const int*   c1    = (const int*)d_in[34];
    float* out = (float*)d_out;

    // ---- workspace carve (with staged reuse) ----
    char* base = (char*)d_ws;
    size_t off = 0;
    auto carve = [&](size_t bytes) -> char* {
        char* p = base + off;
        off += (bytes + 255) & ~(size_t)255;
        return p;
    };
    float* regA = (float*)carve((size_t)NN * DD * 4);   // t0, later: t2 | hmlp2 | h3
    float* regB = (float*)carve((size_t)NN * DD * 4);   // hmlp0, later: re2 | ce2
    float* h    = (float*)carve((size_t)NN * DD * 4);
    float* px   = (float*)carve((size_t)NN0 * DD * 4);  // -> h2, later: gx | z | zz | lw1p
    float* ox   = (float*)carve((size_t)NN0 * DD * 4);
    float* px2  = (float*)carve((size_t)NN1 * DD * 4);
    float* dis0 = (float*)carve((size_t)NN * 4);
    float* dis2 = (float*)carve((size_t)NN1 * 4);
    int*   pb   = (int*)carve((size_t)NN0 * 4);
    int*   pb2  = (int*)carve((size_t)NN1 * 4);

    float* t0    = regA;
    float* t2    = regA;                               // 25.6 MB
    float* hmlp2 = regA + (size_t)NN1 * HH;            // +12.8 MB
    float* h3    = hmlp2 + (size_t)NN1 * DD;           // +12.8 MB
    float* hmlp0 = regB;
    int*   re2   = (int*)regB;                         // after hmlp0 is dead
    int*   ce2   = re2 + EE;
    float* gx    = px;                                 // after h2 (px) is dead
    float* zbuf  = px + 1 * 262144;
    float* zz    = px + 2 * 262144;
    float* lw1p  = px + 3 * 262144;

    const dim3 T(256);
    auto nb = [](long n) { return dim3((unsigned)((n + 255) / 256)); };
    const dim3 FILLG(2048);

    // ===== mpnn0 normalization =====
    k_fill_f32<<<FILLG, T, 0, stream>>>(dis0, 1.0f, (long)NN);
    k_deg<<<nb(EE), T, 0, stream>>>(ei, dis0, EE);
    k_rsqrt<<<nb(NN), T, 0, stream>>>(dis0, NN);

    // ===== mpnn0 node MLP (WMMA, bf16 inputs / f32 accum) =====
    wmma_linear<DD><<<dim3((NN + 127) / 128), T, 0, stream>>>(x,  w0a, b0a, g0, be0, t0,    NN, DD, 1);
    wmma_linear<DD><<<dim3((NN + 127) / 128), T, 0, stream>>>(t0, w0b, b0b, nullptr, nullptr, hmlp0, NN, DD, 0);

    // ===== mpnn0 aggregation (+ residual fused into self-loop init) =====
    k_self_init<<<nb((long)NN * 32), T, 0, stream>>>(hmlp0, x, dis0, h, NN);
    k_edge_scatter<<<dim3((EE + 7) / 8), T, 0, stream>>>(ei, ei + EE, eatt, dis0, hmlp0, h, EE);

    // ===== graclus pool #1 =====
    k_fill_f32<<<FILLG, T, 0, stream>>>(px, -__builtin_inff(), (long)NN0 * DD);
    k_fill_f32<<<FILLG, T, 0, stream>>>(ox, -__builtin_inff(), (long)NN0 * DD);
    k_pool_max_f<<<nb((long)NN * DD), T, 0, stream>>>(h, c0, px, NN, 1);
    k_pool_max_f<<<nb((long)NN * DD), T, 0, stream>>>(x, c0, ox, NN, 0);
    k_combine<<<nb((long)NN0 * DD), T, 0, stream>>>(px, ox, (long)NN0 * DD);   // px <- h2
    k_fill_i32<<<FILLG, T, 0, stream>>>(pb, 0, (long)NN0);
    k_pool_max_i<<<nb(NN), T, 0, stream>>>(batch, c0, pb, NN);

    // ===== graclus pool #2 =====
    k_fill_f32<<<FILLG, T, 0, stream>>>(px2, -__builtin_inff(), (long)NN1 * DD);
    k_pool_max_f<<<nb((long)NN0 * DD), T, 0, stream>>>(px, c1, px2, NN0, 1);
    k_fixneg<<<nb((long)NN1 * DD), T, 0, stream>>>(px2, (long)NN1 * DD);
    k_fill_i32<<<FILLG, T, 0, stream>>>(pb2, 0, (long)NN1);
    k_pool_max_i<<<nb(NN0), T, 0, stream>>>(pb, c1, pb2, NN0);

    // ===== mpnn2: mapped edges, normalization =====
    k_map_edges<<<nb(EE), T, 0, stream>>>(ei, c0, c1, re2, ce2, EE);
    k_fill_f32<<<FILLG, T, 0, stream>>>(dis2, 1.0f, (long)NN1);
    k_deg<<<nb(EE), T, 0, stream>>>(re2, dis2, EE);
    k_rsqrt<<<nb(NN1), T, 0, stream>>>(dis2, NN1);

    // ===== mpnn2 node MLP (128 -> 256 -> 128) =====
    wmma_linear<HH><<<dim3((NN1 + 127) / 128), T, 0, stream>>>(px2, w2a, b2a, g2, be2, t2,    NN1, DD, 1);
    wmma_linear<DD><<<dim3((NN1 + 127) / 128), T, 0, stream>>>(t2,  w2b, b2b, nullptr, nullptr, hmlp2, NN1, HH, 0);

    // ===== mpnn2 aggregation =====
    k_self_init<<<nb((long)NN1 * 32), T, 0, stream>>>(hmlp2, nullptr, dis2, h3, NN1);
    k_edge_scatter<<<dim3((EE + 7) / 8), T, 0, stream>>>(re2, ce2, eatt, dis2, hmlp2, h3, EE);

    // ===== global max pool (lrelu folded: seg_max(lrelu(.)) ) =====
    k_fill_f32<<<FILLG, T, 0, stream>>>(gx, -__builtin_inff(), (long)GG * DD);
    k_pool_max_f<<<nb((long)NN1 * DD), T, 0, stream>>>(h3, pb2, gx, NN1, 1);

    // ===== head =====
    k_pad_w<<<nb(128 * 160), T, 0, stream>>>(lw1, lw1p);
    k_build_z<<<nb(GG * 160), T, 0, stream>>>(gx, en, embw, embb, embg, embe, zbuf);
    wmma_linear<DD><<<dim3(1), T, 0, stream>>>(zbuf, lw1p, lb1, lg, lbe, zz, GG, 160, 1);
    k_head2<<<dim3(1), dim3(64), 0, stream>>>(zz, lw2, lb2, out);
}
